// GAT_67448166416665
// MI455X (gfx1250) — matrix operand order repaired and checked by reading, not compiled, not run on hardware
//
#include <hip/hip_runtime.h>
#include <hip/hip_fp16.h>

typedef __attribute__((ext_vector_type(16))) _Float16 v16h;
typedef __attribute__((ext_vector_type(8)))  _Float16 v8h;
typedef __attribute__((ext_vector_type(8)))  float    v8f;
typedef __attribute__((ext_vector_type(4)))  unsigned u32x4;
typedef __attribute__((ext_vector_type(8)))  int      i32x8;
typedef __attribute__((ext_vector_type(4)))  int      i32x4;

#define BSZ 2
#define NN  2048
#define FIN 128
#define NH1 8
#define NF1 64
#define CH1 (NH1*NF1)   /* 512 */
#define NH2 1
#define NF2 64
#define NWRD (NN/32)    /* 64 mask words per row */

#if defined(__HIP_DEVICE_COMPILE__) && __has_builtin(__builtin_amdgcn_tensor_load_to_lds) && __has_builtin(__builtin_amdgcn_s_wait_tensorcnt)
#define USE_TDM 1
#else
#define USE_TDM 0
#endif

union V16U { v16h v; uint4 u[2]; };
union V8U  { v8h  v; uint4 u; };

static __device__ __forceinline__ float leaky(float v) { return fmaxf(v, 0.2f * v); }

// ---------------- elementwise f32 -> f16 convert ----------------
__global__ void k_cvt_f16(const float* __restrict__ src, _Float16* __restrict__ dst, int n) {
    int i = blockIdx.x * blockDim.x + threadIdx.x;
    if (i < n) dst[i] = (_Float16)src[i];
}

// ---------------- pack adjacency (f32 0/1) into bitmask ----------------
__global__ void k_pack_adj(const float* __restrict__ adj, unsigned* __restrict__ bits, int total) {
    int i = blockIdx.x * blockDim.x + threadIdx.x;   // element over BS*N*N
    if (i >= total) return;
    unsigned m = (unsigned)__ballot(adj[i] != 0.0f); // wave32 -> low 32 bits
    if ((threadIdx.x & 31) == 0) bits[i >> 5] = m;
}

// ---------------- WMMA GEMM: C[M,Nc] = A[M,K](f16,row) * B[Nc,K](f16,row)^T ---------
// mode 0: store f32 row-major [M][Nc]   (skip path)
// mode 1: store f16 head-transposed [b][h][f][n]  (proj path, n = row % NN)
__global__ void k_gemm(const _Float16* __restrict__ A, const _Float16* __restrict__ B,
                       int M, int K, int Nc,
                       float* __restrict__ outF32, _Float16* __restrict__ outT16,
                       int H, int F, int mode) {
    int lane = threadIdx.x & 31;
    int wave = blockIdx.x * (blockDim.x >> 5) + (threadIdx.x >> 5);
    int colTiles = Nc >> 4;
    int tiles = (M >> 4) * colTiles;
    if (wave >= tiles) return;
    int rowT = wave / colTiles, colT = wave % colTiles;
    bool hi = lane >= 16;
    int r = (rowT << 4) + (lane & 15);
    int c = (colT << 4) + (lane & 15);
    // A layout: lanes 0-15 hold K {0..7,16..23}; lanes 16-31 hold K {8..15,24..31}
    const _Float16* aptr = A + (size_t)r * K + (hi ? 8 : 0);
    // B layout: lanes 0-15 hold K {0..7,16..23}; lanes 16-31 hold K {8..15,24..31}
    const _Float16* bptr = B + (size_t)c * K + (hi ? 8 : 0);
    v8f acc = {};
    for (int kb = 0; kb < K; kb += 32) {
        V16U a, b;
        a.u[0] = *(const uint4*)(aptr + kb);
        a.u[1] = *(const uint4*)(aptr + kb + 16);
        b.u[0] = *(const uint4*)(bptr + kb);
        b.u[1] = *(const uint4*)(bptr + kb + 16);
        acc = __builtin_amdgcn_wmma_f32_16x16x32_f16(false, a.v, false, b.v,
                                                     (short)0, acc, false, false);
    }
    if (mode == 0) {
        int row0 = (rowT << 4) + (hi ? 8 : 0);
        float* o = outF32 + (size_t)row0 * Nc + c;
#pragma unroll
        for (int v = 0; v < 8; ++v) o[(size_t)v * Nc] = acc[v];
    } else {
        int hcol = c / F, f = c % F;
        int bidx = (rowT << 4) / NN;                 // 16-row tiles never straddle batches
        int n0 = ((rowT << 4) % NN) + (hi ? 8 : 0);  // 8 consecutive rows per lane
        V8U pk;
#pragma unroll
        for (int v = 0; v < 8; ++v) pk.v[v] = (_Float16)acc[v];
        *(uint4*)(outT16 + ((size_t)(bidx * H + hcol) * F + f) * NN + n0) = pk.u;
    }
}

// ---------------- attention scores s_src/s_tgt per (b,h,n) ----------------
__global__ void k_scores(const _Float16* __restrict__ projT, const float* __restrict__ a_src,
                         const float* __restrict__ a_tgt, float* __restrict__ ssrc,
                         float* __restrict__ stgt, int H, int F) {
    int i = blockIdx.x * blockDim.x + threadIdx.x;  // over BS*H*N
    if (i >= BSZ * H * NN) return;
    int n = i % NN, bh = i / NN, h = bh % H;
    const _Float16* p = projT + (size_t)bh * F * NN + n;
    float s1 = 0.f, s2 = 0.f;
    for (int f = 0; f < F; ++f) {
        float pv = (float)p[(size_t)f * NN];
        s1 += pv * a_src[h * F + f];
        s2 += pv * a_tgt[h * F + f];
    }
    ssrc[i] = s1; stgt[i] = s2;
}

// ---------------- per-row softmax stats (max, 1/sum) : one wave per row -----------
__global__ void k_stats(const unsigned* __restrict__ bits, const float* __restrict__ ssrc,
                        const float* __restrict__ stgt, float* __restrict__ rowm,
                        float* __restrict__ rowinv, int H) {
    int lane = threadIdx.x & 31;
    int wave = blockIdx.x * (blockDim.x >> 5) + (threadIdx.x >> 5); // over BS*H*N
    if (wave >= BSZ * H * NN) return;
    int row = wave % NN, bh = wave / NN, b = bh / H;
    float srow = ssrc[wave];
    const float* st = stgt + (size_t)bh * NN;
    const unsigned* wrd = bits + ((size_t)b * NN + row) * NWRD;
    float m = -3.0e38f, s = 0.f;
    for (int i = 0; i < NWRD; ++i) {
        unsigned w = wrd[i];
        if ((w >> lane) & 1u) {
            float sc = leaky(srow + st[i * 32 + lane]);
            if (sc > m) { s = s * __expf(m - sc) + 1.f; m = sc; }
            else        { s += __expf(sc - m); }
        }
    }
#pragma unroll
    for (int off = 16; off; off >>= 1) {
        float m2 = __shfl_down(m, off, 32);
        float s2 = __shfl_down(s, off, 32);
        if (m2 > m) { s = s * __expf(m - m2) + s2; m = m2; }
        else        { s += s2 * __expf(m2 - m); }
    }
    if (lane == 0) { rowm[wave] = m; rowinv[wave] = 1.0f / s; }
}

// ---------------- fused attn x V (WMMA): wave = 16 rows x 64 feats -------------------
// Block = 8 waves sharing one (b,h): the 32x64 f16 proj tile per K-step is staged into
// LDS once per block (TDM DMA when available), with a 40-half row stride so the WMMA
// B-fetch ds_load_b128s are bank-conflict free. Double buffered: DMA of tile k+1
// overlaps the exp/softmax VALU work of tile k.
#define LSTRIDE 40
__global__ void k_av(const _Float16* __restrict__ projT, const unsigned* __restrict__ bits,
                     const float* __restrict__ ssrc, const float* __restrict__ stgt,
                     const float* __restrict__ rowm, const float* __restrict__ rowinv,
                     float* __restrict__ out, int H, int outStride) {
    const int F = 64;
    __shared__ __align__(16) _Float16 Bs[2][64 * LSTRIDE];
    int lane = threadIdx.x & 31;
    int wid  = threadIdx.x >> 5;
    int wave = blockIdx.x * (blockDim.x >> 5) + wid;
    int rowTiles = NN >> 4;
    int rt = wave % rowTiles, bh = wave / rowTiles, h = bh % H, b = bh / H;
    bool hi = lane >= 16;
    int r = (rt << 4) + (lane & 15);          // my A-matrix row (node index)
    int rowIdx = bh * NN + r;
    float srow = ssrc[rowIdx];
    float m = rowm[rowIdx];
    float inv = rowinv[rowIdx];
    const float* st = stgt + (size_t)bh * NN;
    const unsigned* wrd = bits + ((size_t)b * NN + r) * NWRD;
    // uniform-per-block head slice for staging
    int bh0 = (blockIdx.x * (blockDim.x >> 5)) / rowTiles;
    const _Float16* pslab = projT + (size_t)bh0 * F * NN;

#if USE_TDM
    auto issue_tile = [&](int kb, int buf) {
        if (wid == 0) {
            unsigned lds_off = (unsigned)(unsigned long long)(void*)&Bs[buf][0];
            unsigned long long ga = (unsigned long long)(pslab + ((size_t)kb << 5));
            u32x4 g0;
            g0[0] = 1u;                                   // count=1, user descriptor
            g0[1] = lds_off;                              // lds_addr
            g0[2] = (unsigned)(ga & 0xffffffffu);         // global_addr lo
            g0[3] = (unsigned)(((ga >> 32) & 0x01ffffffu) | 0x80000000u); // addr hi | type=2
            i32x8 g1;
            // data_size=2B (code1), pad_enable, pad_interval=64B (code3), pad_amount=16B (code3)
            g1[0] = (int)((1u << 16) | (1u << 20) | (3u << 22) | (3u << 25));
            g1[1] = (int)(32u << 16);                     // tensor_dim0 = 32 (lo16 @ [31:16])
            g1[2] = (int)(64u << 16);                     // td0 hi=0 | tensor_dim1 = 64
            g1[3] = (int)(32u << 16);                     // td1 hi=0 | tile_dim0 = 32
            g1[4] = (int)(64u);                           // tile_dim1 = 64, tile_dim2 = 0
            g1[5] = (int)(NN);                            // tensor_dim0_stride lo32 = 2048
            g1[6] = 0;                                    // stride0 hi | stride1 lo
            g1[7] = 0;                                    // stride1 hi
            i32x4 z4 = {0, 0, 0, 0};
            i32x8 z8 = {0, 0, 0, 0, 0, 0, 0, 0};
            __builtin_amdgcn_tensor_load_to_lds(g0, g1, z4, z4, z8, 0);
        }
    };
    issue_tile(0, 0);
#else
    auto stage_tile = [&](int kb, int buf) {
        int t = threadIdx.x;
        int f = t >> 2, chunk = t & 3;                    // 256 threads x 16B = 4KB tile
        uint4 v = *(const uint4*)(pslab + (size_t)f * NN + ((size_t)kb << 5) + chunk * 8);
        *(uint4*)(&Bs[buf][f * LSTRIDE + chunk * 8]) = v;
    };
    stage_tile(0, 0);
#endif

    v8f c0 = {}, c1 = {}, c2 = {}, c3 = {};
    for (int kb = 0; kb < NWRD; ++kb) {
        int cur = kb & 1;
#if USE_TDM
        if (kb + 1 < NWRD) {
            issue_tile(kb + 1, cur ^ 1);
            if (wid == 0) __builtin_amdgcn_s_wait_tensorcnt(1);
        } else {
            if (wid == 0) __builtin_amdgcn_s_wait_tensorcnt(0);
        }
#else
        if (kb + 1 < NWRD) stage_tile(kb + 1, cur ^ 1);
#endif
        __syncthreads();

        unsigned w = wrd[kb];
        float stg = st[kb * 32 + lane];       // cooperative tile of s_tgt
        V16U a;
#pragma unroll
        for (int idx = 0; idx < 16; ++idx) {
            // A layout K index for this slot (contiguous; lanes>=16 shifted by +16)
            int k = idx + (hi ? 16 : 0);
            float sc = __shfl(stg, k, 32);
            float v = leaky(srow + sc);
            float wgt = ((w >> k) & 1u) ? __expf(v - m) * inv : 0.0f;
            a.v[idx] = (_Float16)wgt;
        }
        const _Float16* lp = &Bs[cur][(size_t)(lane & 15) * LSTRIDE + (hi ? 16 : 0)];
        V16U b0, b1, b2, b3;
        b0.u[0] = *(const uint4*)(lp);                 b0.u[1] = *(const uint4*)(lp + 8);
        b1.u[0] = *(const uint4*)(lp + 16 * LSTRIDE);  b1.u[1] = *(const uint4*)(lp + 16 * LSTRIDE + 8);
        b2.u[0] = *(const uint4*)(lp + 32 * LSTRIDE);  b2.u[1] = *(const uint4*)(lp + 32 * LSTRIDE + 8);
        b3.u[0] = *(const uint4*)(lp + 48 * LSTRIDE);  b3.u[1] = *(const uint4*)(lp + 48 * LSTRIDE + 8);
        c0 = __builtin_amdgcn_wmma_f32_16x16x32_f16(false, a.v, false, b0.v, (short)0, c0, false, false);
        c1 = __builtin_amdgcn_wmma_f32_16x16x32_f16(false, a.v, false, b1.v, (short)0, c1, false, false);
        c2 = __builtin_amdgcn_wmma_f32_16x16x32_f16(false, a.v, false, b2.v, (short)0, c2, false, false);
        c3 = __builtin_amdgcn_wmma_f32_16x16x32_f16(false, a.v, false, b3.v, (short)0, c3, false, false);
        __syncthreads();                       // protect buffer reuse (double buffer)
    }
    int n0 = (rt << 4) + (hi ? 8 : 0);
    float* o = out + ((size_t)b * NN + n0) * outStride + (size_t)h * F + (lane & 15);
#pragma unroll
    for (int v = 0; v < 8; ++v) {
        float* ro = o + (size_t)v * outStride;
        ro[0] = c0[v]; ro[16] = c1[v]; ro[32] = c2[v]; ro[48] = c3[v];
    }
}

// ---------------- epilogues ----------------
__global__ void k_epi1(float* __restrict__ hbuf, const float* __restrict__ skip,
                       const float* __restrict__ bias, _Float16* __restrict__ h16) {
    int i = blockIdx.x * blockDim.x + threadIdx.x;
    if (i >= BSZ * NN * CH1) return;
    float v = hbuf[i] + skip[i] + bias[i % CH1];
    float e = v > 0.f ? v : (__expf(v) - 1.f);   // ELU(alpha=1)
    hbuf[i] = e;
    h16[i] = (_Float16)e;
}

__global__ void k_epi2(float* __restrict__ outp, const float* __restrict__ skip,
                       const float* __restrict__ bias) {
    int i = blockIdx.x * blockDim.x + threadIdx.x;
    if (i >= BSZ * NN * NF2) return;
    outp[i] = outp[i] + skip[i] + bias[i % NF2];
}

extern "C" void kernel_launch(void* const* d_in, const int* in_sizes, int n_in,
                              void* d_out, int out_size, void* d_ws, size_t ws_size,
                              hipStream_t stream) {
    (void)in_sizes; (void)n_in; (void)out_size; (void)ws_size;
    const float* x      = (const float*)d_in[0];
    const float* adj    = (const float*)d_in[1];
    const float* W1     = (const float*)d_in[2];
    const float* a_src1 = (const float*)d_in[3];
    const float* a_tgt1 = (const float*)d_in[4];
    const float* Wsk1   = (const float*)d_in[5];
    const float* b1     = (const float*)d_in[6];
    const float* W2     = (const float*)d_in[7];
    const float* a_src2 = (const float*)d_in[8];
    const float* a_tgt2 = (const float*)d_in[9];
    const float* Wsk2   = (const float*)d_in[10];
    const float* b2     = (const float*)d_in[11];
    float* outp = (float*)d_out;

    char* ws = (char*)d_ws;
    size_t off = 0;
    auto alloc = [&](size_t bytes) -> char* {
        char* p = ws + off; off += (bytes + 255) & ~(size_t)255; return p;
    };
    unsigned* adjbits = (unsigned*)alloc((size_t)BSZ * NN * NWRD * 4);
    _Float16* x16     = (_Float16*)alloc((size_t)BSZ * NN * FIN * 2);
    _Float16* W1h     = (_Float16*)alloc((size_t)CH1 * FIN * 2);
    _Float16* Wsk1h   = (_Float16*)alloc((size_t)CH1 * FIN * 2);
    _Float16* W2h     = (_Float16*)alloc((size_t)NF2 * CH1 * 2);
    _Float16* Wsk2h   = (_Float16*)alloc((size_t)NF2 * CH1 * 2);
    _Float16* projT1  = (_Float16*)alloc((size_t)BSZ * NH1 * NF1 * NN * 2);
    float*    skip1   = (float*)   alloc((size_t)BSZ * NN * CH1 * 4);
    float*    ssrc1   = (float*)   alloc((size_t)BSZ * NH1 * NN * 4);
    float*    stgt1   = (float*)   alloc((size_t)BSZ * NH1 * NN * 4);
    float*    rowm1   = (float*)   alloc((size_t)BSZ * NH1 * NN * 4);
    float*    rinv1   = (float*)   alloc((size_t)BSZ * NH1 * NN * 4);
    float*    hbuf    = (float*)   alloc((size_t)BSZ * NN * CH1 * 4);
    _Float16* h16     = (_Float16*)alloc((size_t)BSZ * NN * CH1 * 2);
    _Float16* projT2  = (_Float16*)alloc((size_t)BSZ * NH2 * NF2 * NN * 2);
    float*    skip2   = (float*)   alloc((size_t)BSZ * NN * NF2 * 4);
    float*    ssrc2   = (float*)   alloc((size_t)BSZ * NH2 * NN * 4);
    float*    stgt2   = (float*)   alloc((size_t)BSZ * NH2 * NN * 4);
    float*    rowm2   = (float*)   alloc((size_t)BSZ * NH2 * NN * 4);
    float*    rinv2   = (float*)   alloc((size_t)BSZ * NH2 * NN * 4);

    const int TB = 256;
    auto grid = [](int n, int tb) { return (n + tb - 1) / tb; };

    // conversions
    k_cvt_f16<<<grid(BSZ*NN*FIN, TB), TB, 0, stream>>>(x, x16, BSZ*NN*FIN);
    k_cvt_f16<<<grid(CH1*FIN, TB), TB, 0, stream>>>(W1, W1h, CH1*FIN);
    k_cvt_f16<<<grid(CH1*FIN, TB), TB, 0, stream>>>(Wsk1, Wsk1h, CH1*FIN);
    k_cvt_f16<<<grid(NF2*CH1, TB), TB, 0, stream>>>(W2, W2h, NF2*CH1);
    k_cvt_f16<<<grid(NF2*CH1, TB), TB, 0, stream>>>(Wsk2, Wsk2h, NF2*CH1);
    // adjacency bitmask
    k_pack_adj<<<grid(BSZ*NN*NN, TB), TB, 0, stream>>>(adj, adjbits, BSZ*NN*NN);

    // ---------------- layer 1 ----------------
    {
        int tiles = (BSZ*NN/16) * (CH1/16);                 // 8192 wave-tiles
        k_gemm<<<grid(tiles*32, TB), TB, 0, stream>>>(x16, W1h, BSZ*NN, FIN, CH1,
                                                      nullptr, projT1, NH1, NF1, 1);
        k_gemm<<<grid(tiles*32, TB), TB, 0, stream>>>(x16, Wsk1h, BSZ*NN, FIN, CH1,
                                                      skip1, nullptr, NH1, NF1, 0);
        k_scores<<<grid(BSZ*NH1*NN, TB), TB, 0, stream>>>(projT1, a_src1, a_tgt1,
                                                          ssrc1, stgt1, NH1, NF1);
        k_stats<<<grid(BSZ*NH1*NN*32, TB), TB, 0, stream>>>(adjbits, ssrc1, stgt1,
                                                            rowm1, rinv1, NH1);
        int waves = BSZ * NH1 * (NN/16);                    // 2048
        k_av<<<grid(waves*32, TB), TB, 0, stream>>>(projT1, adjbits, ssrc1, stgt1,
                                                    rowm1, rinv1, hbuf, NH1, CH1);
        k_epi1<<<grid(BSZ*NN*CH1, TB), TB, 0, stream>>>(hbuf, skip1, b1, h16);
    }

    // ---------------- layer 2 ----------------
    {
        int tiles = (BSZ*NN/16) * (NF2/16);                 // 1024 wave-tiles
        k_gemm<<<grid(tiles*32, TB), TB, 0, stream>>>(h16, W2h, BSZ*NN, CH1, NF2,
                                                      nullptr, projT2, NH2, NF2, 1);
        k_gemm<<<grid(tiles*32, TB), TB, 0, stream>>>(h16, Wsk2h, BSZ*NN, CH1, NF2,
                                                      skip2, nullptr, NH2, NF2, 0);
        k_scores<<<grid(BSZ*NH2*NN, TB), TB, 0, stream>>>(projT2, a_src2, a_tgt2,
                                                          ssrc2, stgt2, NH2, NF2);
        k_stats<<<grid(BSZ*NH2*NN*32, TB), TB, 0, stream>>>(adjbits, ssrc2, stgt2,
                                                            rowm2, rinv2, NH2);
        int waves = BSZ * NH2 * (NN/16);                    // 256
        k_av<<<grid(waves*32, TB), TB, 0, stream>>>(projT2, adjbits, ssrc2, stgt2,
                                                    rowm2, rinv2, outp, NH2, NF2);
        k_epi2<<<grid(BSZ*NN*NF2, TB), TB, 0, stream>>>(outp, skip2, b2);
    }
}
